// GATv2Model_63891933496079
// MI455X (gfx1250) — compile-verified, matrix operand order checked by reading, not run on hardware
//
#include <hip/hip_runtime.h>
#include <hip/hip_bf16.h>

typedef __attribute__((ext_vector_type(16))) __bf16 v16bf;
typedef __attribute__((ext_vector_type(8)))  __bf16 v8bf;
typedef __attribute__((ext_vector_type(8)))  float  v8f;

#define HDIM 128

#define SHUF16(a, b) __builtin_shufflevector(a, b, 0, 1, 2, 3, 4, 5, 6, 7, 8, 9, 10, 11, 12, 13, 14, 15)

__device__ __forceinline__ float wave_sum(float x) {
#pragma unroll
  for (int o = 16; o > 0; o >>= 1) x += __shfl_xor(x, o, 32);
  return x;
}

// order-preserving f32 <-> u32 encoding for atomicMax on floats
__device__ __forceinline__ unsigned f32_order(float x) {
  unsigned u = __float_as_uint(x);
  return (u & 0x80000000u) ? ~u : (u | 0x80000000u);
}
__device__ __forceinline__ float f32_unorder(unsigned u) {
  return __uint_as_float((u & 0x80000000u) ? (u ^ 0x80000000u) : ~u);
}

// -----------------------------------------------------------------------------
// Operand packing: A operands row-major f32 -> bf16; W -> transposed bf16
// (Wt[n*K+k] = W[k*Nc+n]) so the B fragment load pattern is identical to A's:
// per lane, per k-step: 8 contiguous bf16 (16B) from lo half + 16B from hi half.
// -----------------------------------------------------------------------------
__global__ void pack_bf16(const float* __restrict__ s, __bf16* __restrict__ d, int n4) {
  int i = blockIdx.x * blockDim.x + threadIdx.x;
  if (i >= n4) return;
  float4 v = *(const float4*)(s + (size_t)i * 4);
  __bf16* p = d + (size_t)i * 4;
  p[0] = (__bf16)v.x; p[1] = (__bf16)v.y; p[2] = (__bf16)v.z; p[3] = (__bf16)v.w;
}

__global__ void pack_wT_bf16(const float* __restrict__ W, __bf16* __restrict__ Wt,
                             int K, int Nc) {
  int idx = blockIdx.x * blockDim.x + threadIdx.x;
  if (idx >= K * Nc) return;
  int n = idx / K, k = idx % K;
  Wt[idx] = (__bf16)W[k * Nc + n];
}

// -----------------------------------------------------------------------------
// O[M,NC] = act(A[M,K] @ W + bias). A: bf16 row-major. Wt: bf16 [NC,K].
// Block = 256 = 8 waves, one 16x16 tile per wave; K fully unrolled.
// ISA 7.12.2 16-bit fragment: lane half selects K lo/hi 8-groups.
// -----------------------------------------------------------------------------
template <int K, int NC>
__global__ __launch_bounds__(256) void gemm_wmma(
    const __bf16* __restrict__ Ab, const __bf16* __restrict__ Wt,
    const float* __restrict__ bias, float* __restrict__ O, int M, int act) {
  constexpr int colTiles = NC / 16;
  constexpr int rtpb = 8 / colTiles;
  const int lane = threadIdx.x & 31;
  const int wave = threadIdx.x >> 5;
  const int rowTile = blockIdx.x * rtpb + wave / colTiles;
  const int colTile = wave % colTiles;
  if (rowTile * 16 >= M) return;  // wave-uniform exit (EXEC all-ones at WMMA)
  const int half = lane >> 4;
  const int lrow = lane & 15;
  const int rowBase = rowTile * 16;
  const int col = colTile * 16 + lrow;
  const v8bf* __restrict__ Arow = (const v8bf*)(Ab + (size_t)(rowBase + lrow) * K);
  const v8bf* __restrict__ Brow = (const v8bf*)(Wt + (size_t)col * K);
  v8f acc = {};
#pragma unroll
  for (int k0 = 0; k0 < K; k0 += 32) {
    const int i0 = (k0 >> 3) + half;
    v16bf a = SHUF16(Arow[i0], Arow[i0 + 2]);
    v16bf b = SHUF16(Brow[i0], Brow[i0 + 2]);
    acc = __builtin_amdgcn_wmma_f32_16x16x32_bf16(false, a, false, b,
                                                  (short)0, acc, false, false);
  }
  const float bv = bias[col];
#pragma unroll
  for (int r = 0; r < 8; ++r) {  // C/D: VGPR r -> row = rowBase + r + 8*half
    float v = acc[r] + bv;
    if (act) v = fmaxf(v, 0.f);
    O[(size_t)(rowBase + r + (half << 3)) * NC + col] = v;
  }
}

// Dual-B GEMM for the per-layer lin_l/lin_r pair: shares the A fragments,
// two WMMAs per k-step.
template <int K, int NC>
__global__ __launch_bounds__(256) void gemm_wmma_dual(
    const __bf16* __restrict__ Ab, const __bf16* __restrict__ Wt0,
    const __bf16* __restrict__ Wt1, const float* __restrict__ bias0,
    const float* __restrict__ bias1, float* __restrict__ O0,
    float* __restrict__ O1, int M) {
  constexpr int colTiles = NC / 16;
  constexpr int rtpb = 8 / colTiles;
  const int lane = threadIdx.x & 31;
  const int wave = threadIdx.x >> 5;
  const int rowTile = blockIdx.x * rtpb + wave / colTiles;
  const int colTile = wave % colTiles;
  if (rowTile * 16 >= M) return;
  const int half = lane >> 4;
  const int lrow = lane & 15;
  const int rowBase = rowTile * 16;
  const int col = colTile * 16 + lrow;
  const v8bf* __restrict__ Arow = (const v8bf*)(Ab + (size_t)(rowBase + lrow) * K);
  const v8bf* __restrict__ B0 = (const v8bf*)(Wt0 + (size_t)col * K);
  const v8bf* __restrict__ B1 = (const v8bf*)(Wt1 + (size_t)col * K);
  v8f acc0 = {}, acc1 = {};
#pragma unroll
  for (int k0 = 0; k0 < K; k0 += 32) {
    const int i0 = (k0 >> 3) + half;
    v16bf a  = SHUF16(Arow[i0], Arow[i0 + 2]);
    v16bf b0 = SHUF16(B0[i0], B0[i0 + 2]);
    v16bf b1 = SHUF16(B1[i0], B1[i0 + 2]);
    acc0 = __builtin_amdgcn_wmma_f32_16x16x32_bf16(false, a, false, b0,
                                                   (short)0, acc0, false, false);
    acc1 = __builtin_amdgcn_wmma_f32_16x16x32_bf16(false, a, false, b1,
                                                   (short)0, acc1, false, false);
  }
  const float bv0 = bias0[col], bv1 = bias1[col];
#pragma unroll
  for (int r = 0; r < 8; ++r) {
    const size_t o = (size_t)(rowBase + r + (half << 3)) * NC + col;
    O0[o] = acc0[r] + bv0;
    O1[o] = acc1[r] + bv1;
  }
}

// -----------------------------------------------------------------------------
__global__ void zero_kernel(float* __restrict__ p, int n) {
  int i = blockIdx.x * blockDim.x + threadIdx.x;
  if (i < n) p[i] = 0.f;
}

// one wave per edge: s[e] = att . leaky_relu(xl[src] + xr[dst]); segment max
__global__ void edge_score(const int* __restrict__ src, const int* __restrict__ dst,
                           const float* __restrict__ xl, const float* __restrict__ xr,
                           const float* __restrict__ att, float* __restrict__ s,
                           unsigned* __restrict__ smax, int Etot, int Ereal) {
  int wid = (int)((blockIdx.x * blockDim.x + threadIdx.x) >> 5);
  int lane = threadIdx.x & 31;
  if (wid >= Etot) return;
  int sI, dI;
  if (wid < Ereal) { sI = src[wid]; dI = dst[wid]; } else { sI = dI = wid - Ereal; }
  float acc = 0.f;
#pragma unroll
  for (int j = 0; j < 4; ++j) {
    int f = lane + (j << 5);
    float v = xl[sI * HDIM + f] + xr[dI * HDIM + f];
    v = (v > 0.f) ? v : 0.2f * v;  // LeakyReLU(0.2)
    acc += v * att[f];
  }
  acc = wave_sum(acc);
  if (lane == 0) {
    s[wid] = acc;
    atomicMax(&smax[dI], f32_order(acc));
  }
}

// w[e] = exp(s[e] - max[dst]); z[dst] += w[e]
__global__ void edge_exp(const int* __restrict__ dst, const float* __restrict__ s,
                         const unsigned* __restrict__ smax, float* __restrict__ w,
                         float* __restrict__ z, int Etot, int Ereal) {
  int e = blockIdx.x * blockDim.x + threadIdx.x;
  if (e >= Etot) return;
  int dI = (e < Ereal) ? dst[e] : (e - Ereal);
  float wv = __expf(s[e] - f32_unorder(smax[dI]));
  w[e] = wv;
  atomicAdd(&z[dI], wv);
}

// agg[dst] += xl[src] * alpha (alpha = w / z[dst]); one wave per edge
__global__ void edge_aggr(const int* __restrict__ src, const int* __restrict__ dst,
                          const float* __restrict__ xl, const float* __restrict__ w,
                          const float* __restrict__ z, float* __restrict__ agg,
                          int Etot, int Ereal) {
  int wid = (int)((blockIdx.x * blockDim.x + threadIdx.x) >> 5);
  int lane = threadIdx.x & 31;
  if (wid >= Etot) return;
  int sI, dI;
  if (wid < Ereal) { sI = src[wid]; dI = dst[wid]; } else { sI = dI = wid - Ereal; }
  float alpha = w[wid] / z[dI];
#pragma unroll
  for (int j = 0; j < 4; ++j) {
    int f = lane + (j << 5);
    atomicAdd(&agg[dI * HDIM + f], xl[sI * HDIM + f] * alpha);
  }
}

// one wave per node: h = [relu](LayerNorm(agg + bg [+ h_residual]))  (in-place)
__global__ void node_post(float* __restrict__ h, const float* __restrict__ agg,
                          const float* __restrict__ bg, const float* __restrict__ g,
                          const float* __restrict__ bln, int Nn, int addRes, int doRelu) {
  int wid = (int)((blockIdx.x * blockDim.x + threadIdx.x) >> 5);
  int lane = threadIdx.x & 31;
  if (wid >= Nn) return;
  float v[4];
  float sum = 0.f;
#pragma unroll
  for (int j = 0; j < 4; ++j) {
    int f = lane + (j << 5);
    float x = agg[wid * HDIM + f] + bg[f];
    if (addRes) x += h[wid * HDIM + f];
    v[j] = x; sum += x;
  }
  float mu = wave_sum(sum) * (1.f / HDIM);
  float var = 0.f;
#pragma unroll
  for (int j = 0; j < 4; ++j) { float d = v[j] - mu; var += d * d; }
  var = wave_sum(var) * (1.f / HDIM);
  float rs = rsqrtf(var + 1e-5f);
#pragma unroll
  for (int j = 0; j < 4; ++j) {
    int f = lane + (j << 5);
    float y = (v[j] - mu) * rs * g[f] + bln[f];
    if (doRelu) y = fmaxf(y, 0.f);
    h[wid * HDIM + f] = y;
  }
}

// final tiny GEMM: out[N,2] = t[N,64] @ W2[64,2] + b2
__global__ void mlp2_kernel(const float* __restrict__ t, const float* __restrict__ W2,
                            const float* __restrict__ b2, float* __restrict__ out, int M) {
  int n = blockIdx.x * blockDim.x + threadIdx.x;
  if (n >= M) return;
  float o0 = b2[0], o1 = b2[1];
#pragma unroll 16
  for (int k = 0; k < 64; ++k) {
    float v = t[n * 64 + k];
    o0 = fmaf(v, W2[2 * k], o0);
    o1 = fmaf(v, W2[2 * k + 1], o1);
  }
  out[n * 2] = o0;
  out[n * 2 + 1] = o1;
}

// -----------------------------------------------------------------------------
extern "C" void kernel_launch(void* const* d_in, const int* in_sizes, int n_in,
                              void* d_out, int out_size, void* d_ws, size_t ws_size,
                              hipStream_t stream) {
  const float* x    = (const float*)d_in[0];
  const int*   ei   = (const int*)d_in[1];
  const float* Win  = (const float*)d_in[2];
  const float* bin_ = (const float*)d_in[3];
  const float* Wl   = (const float*)d_in[4];
  const float* bl   = (const float*)d_in[5];
  const float* Wr   = (const float*)d_in[6];
  const float* br   = (const float*)d_in[7];
  const float* att  = (const float*)d_in[8];
  const float* bg   = (const float*)d_in[9];
  const float* lng  = (const float*)d_in[10];
  const float* lnb  = (const float*)d_in[11];
  const float* W1   = (const float*)d_in[12];
  const float* b1   = (const float*)d_in[13];
  const float* W2   = (const float*)d_in[14];
  const float* b2   = (const float*)d_in[15];
  float* out = (float*)d_out;

  const int N = in_sizes[0] / 64;  // 50000
  const int E = in_sizes[1] / 2;   // 800000
  const int Etot = E + N;
  const int* src = ei;
  const int* dst = ei + E;

  // ---- workspace carve-out ----
  float* ws = (float*)d_ws;
  size_t off = 0;
  float* h    = ws + off; off += (size_t)N * HDIM;
  float* xl   = ws + off; off += (size_t)N * HDIM;
  float* xr   = ws + off; off += (size_t)N * HDIM;
  float* agg  = ws + off; off += (size_t)N * HDIM;  // agg,z,smax contiguous
  float* z    = ws + off; off += N;
  unsigned* smax = (unsigned*)(ws + off); off += N;
  float* sbuf = ws + off; off += Etot;
  float* wbuf = ws + off; off += Etot;
  float* t = xr;  // MLP hidden reuses xr after the last layer
  // bf16 region (16B-aligned: all prior carves are multiples of 4 floats)
  __bf16* bws = (__bf16*)(ws + off);
  size_t boff = 0;
  __bf16* ab   = bws + boff; boff += (size_t)N * HDIM;  // packed A operand
  __bf16* wbin = bws + boff; boff += 64 * HDIM;
  __bf16* wbl  = bws + boff; boff += 3 * HDIM * HDIM;
  __bf16* wbr  = bws + boff; boff += 3 * HDIM * HDIM;
  __bf16* wb1  = bws + boff; boff += HDIM * 64;

  dim3 blk(256);
  const int gemmGrid128 = (N + 15) / 16;        // NC=128: 1 row tile / block
  const int gemmGrid64  = ((N + 15) / 16 + 1) / 2;  // NC=64: 2 row tiles / block
  const int edgeWaveBlocks = (Etot + 7) / 8;
  const int nodeWaveBlocks = (N + 7) / 8;

  // ---- pack all weights (tiny) ----
  pack_wT_bf16<<<(64 * HDIM + 255) / 256, blk, 0, stream>>>(Win, wbin, 64, HDIM);
  for (int i = 0; i < 3; ++i) {
    pack_wT_bf16<<<(HDIM * HDIM + 255) / 256, blk, 0, stream>>>(
        Wl + (size_t)i * HDIM * HDIM, wbl + (size_t)i * HDIM * HDIM, HDIM, HDIM);
    pack_wT_bf16<<<(HDIM * HDIM + 255) / 256, blk, 0, stream>>>(
        Wr + (size_t)i * HDIM * HDIM, wbr + (size_t)i * HDIM * HDIM, HDIM, HDIM);
  }
  pack_wT_bf16<<<(HDIM * 64 + 255) / 256, blk, 0, stream>>>(W1, wb1, HDIM, 64);

  // ---- input projection + ReLU ----
  pack_bf16<<<(N * 64 / 4 + 255) / 256, blk, 0, stream>>>(x, ab, N * 64 / 4);
  gemm_wmma<64, 128><<<gemmGrid128, blk, 0, stream>>>(ab, wbin, bin_, h, N, 1);

  // ---- GATv2 layers ----
  for (int i = 0; i < 3; ++i) {
    pack_bf16<<<(N * HDIM / 4 + 255) / 256, blk, 0, stream>>>(h, ab, N * HDIM / 4);
    gemm_wmma_dual<128, 128><<<gemmGrid128, blk, 0, stream>>>(
        ab, wbl + (size_t)i * HDIM * HDIM, wbr + (size_t)i * HDIM * HDIM,
        bl + i * HDIM, br + i * HDIM, xl, xr, N);
    const int zc = N * HDIM + 2 * N;  // agg + z + smax
    zero_kernel<<<(zc + 255) / 256, blk, 0, stream>>>(agg, zc);
    edge_score<<<edgeWaveBlocks, blk, 0, stream>>>(src, dst, xl, xr, att + i * HDIM,
                                                   sbuf, smax, Etot, E);
    edge_exp<<<(Etot + 255) / 256, blk, 0, stream>>>(dst, sbuf, smax, wbuf, z, Etot, E);
    edge_aggr<<<edgeWaveBlocks, blk, 0, stream>>>(src, dst, xl, wbuf, z, agg, Etot, E);
    node_post<<<nodeWaveBlocks, blk, 0, stream>>>(h, agg, bg + i * HDIM, lng + i * HDIM,
                                                  lnb + i * HDIM, N, (i > 0) ? 1 : 0,
                                                  (i < 2) ? 1 : 0);
  }

  // ---- output MLP ----
  pack_bf16<<<(N * HDIM / 4 + 255) / 256, blk, 0, stream>>>(h, ab, N * HDIM / 4);
  gemm_wmma<128, 64><<<gemmGrid64, blk, 0, stream>>>(ab, wb1, b1, t, N, 1);
  mlp2_kernel<<<(N + 255) / 256, blk, 0, stream>>>(t, W2, b2, out, N);
}
